// TransformerBlock_61538291417743
// MI455X (gfx1250) — matrix operand order repaired and checked by reading, not compiled, not run on hardware
//
#include <hip/hip_runtime.h>
#include <math.h>

// ---------------- problem constants ----------------
#define DIM      512
#define HEADS    8
#define HEAD_DIM 64
#define WIN      3
#define NTOK     9
#define HIDN     2048
#define NWIN     8192
#define TOKENS   (NWIN * NTOK)   // 73728
#define ATTN_SCALE 0.125f        // HEAD_DIM^-0.5
#define LN_EPS   1e-5f

typedef __bf16 bf16_t;
typedef __attribute__((ext_vector_type(16))) __bf16 bf16x16;
typedef __attribute__((ext_vector_type(8)))  float  floatx8;

union FragAB { bf16x16 v; uint4 q[2]; };

// ---------------- LayerNorm (f32 in -> bf16 out), one wave32 per token ----------------
__global__ __launch_bounds__(256)
void ln_bf16_kernel(const float* __restrict__ x,
                    const float* __restrict__ gamma,
                    const float* __restrict__ beta,
                    bf16_t* __restrict__ out) {
  const int wave = threadIdx.x >> 5;
  const int lane = threadIdx.x & 31;
  const int tok  = blockIdx.x * 8 + wave;
  const float* px = x + (size_t)tok * DIM;
  float v[16];
  float s = 0.f, s2 = 0.f;
#pragma unroll
  for (int i = 0; i < 16; ++i) {
    v[i] = px[i * 32 + lane];
    s  += v[i];
    s2 += v[i] * v[i];
  }
#pragma unroll
  for (int m = 16; m >= 1; m >>= 1) {
    s  += __shfl_xor(s,  m, 32);
    s2 += __shfl_xor(s2, m, 32);
  }
  const float mu   = s  * (1.0f / DIM);
  const float var  = s2 * (1.0f / DIM) - mu * mu;
  const float rstd = rsqrtf(var + LN_EPS);
  bf16_t* po = out + (size_t)tok * DIM;
#pragma unroll
  for (int i = 0; i < 16; ++i) {
    const int c = i * 32 + lane;
    po[c] = (bf16_t)((v[i] - mu) * rstd * gamma[c] + beta[c]);
  }
}

// ---------------- weight transpose + f32->bf16: out[k*N+n] = W[n*K+k] ----------------
__global__ void cvt_wT_kernel(const float* __restrict__ W, bf16_t* __restrict__ out,
                              int N, int K) {
  const int idx = blockIdx.x * blockDim.x + threadIdx.x;
  if (idx >= N * K) return;
  const int k = idx / N;
  const int n = idx % N;
  out[idx] = (bf16_t)W[(size_t)n * K + k];
}

// ---------------- tiled bf16 WMMA GEMM: C = A(MxK) * B(KxN) + epilogue ----------------
// Double-buffered LDS, one barrier per K-iteration; global loads for tile k+1
// are issued before the WMMAs of tile k so load latency hides under compute.
// EPI 0: +bias -> bf16 out               (QKV)
// EPI 1: +bias, exact GELU -> bf16 out   (FC1)
// EPI 2: +bias +resid -> f32 out         (proj / FC2)
#define BM 128
#define BN 128
#define BK 32

template <int EPI>
__global__ __launch_bounds__(256)
void gemm_wmma_kernel(const bf16_t* __restrict__ A, const bf16_t* __restrict__ Bm,
                      const float* __restrict__ bias, const float* __restrict__ resid,
                      bf16_t* __restrict__ outB, float* __restrict__ outF,
                      int M, int N, int K) {
  __shared__ bf16_t sA[2][BM][BK + 8];   // +8 bf16 pad, row stride 80B (16B aligned)
  __shared__ bf16_t sB[2][BK][BN + 8];   // row stride 272B (16B aligned)

  const int tid  = threadIdx.x;
  const int wave = tid >> 5;
  const int lane = tid & 31;
  const int wm   = wave >> 1;          // 0..3 -> M sub-tile
  const int wn   = wave & 1;           // 0..1 -> N sub-tile
  const int row0 = blockIdx.y * BM;
  const int col0 = blockIdx.x * BN;

  // global->LDS staging map (each thread: 16 bf16 of A and 16 bf16 of B)
  const int arow = tid >> 1;           // 0..127
  const int acol = (tid & 1) * 16;     // 0 / 16
  const int brow = tid >> 3;           // 0..31
  const int bcol = (tid & 7) * 16;     // 0..112

  const floatx8 vzero = {0.f, 0.f, 0.f, 0.f, 0.f, 0.f, 0.f, 0.f};
  floatx8 acc[2][4];
#pragma unroll
  for (int i = 0; i < 2; ++i)
#pragma unroll
    for (int j = 0; j < 4; ++j) acc[i][j] = vzero;

  const int lh  = lane >> 4;           // lane half (A frag K-group select)
  const int l16 = lane & 15;

  // ---- prologue: stage K-tile 0 into buffer 0 ----
  {
    const uint4* gA = reinterpret_cast<const uint4*>(A + (size_t)(row0 + arow) * K + acol);
    const uint4* gB = reinterpret_cast<const uint4*>(Bm + (size_t)brow * N + col0 + bcol);
    uint4 ra0 = gA[0], ra1 = gA[1];
    uint4 rb0 = gB[0], rb1 = gB[1];
    *reinterpret_cast<uint4*>(&sA[0][arow][acol])     = ra0;
    *reinterpret_cast<uint4*>(&sA[0][arow][acol + 8]) = ra1;
    *reinterpret_cast<uint4*>(&sB[0][brow][bcol])     = rb0;
    *reinterpret_cast<uint4*>(&sB[0][brow][bcol + 8]) = rb1;
  }

  int buf = 0;
  for (int k0 = 0; k0 < K; k0 += BK) {
    __syncthreads();   // buffer `buf` fully staged; alternate buffer free

    const bool has_next = (k0 + BK) < K;
    uint4 ra0, ra1, rb0, rb1;
    if (has_next) {    // issue next tile's global loads early (latency hides under WMMA)
      const uint4* gA = reinterpret_cast<const uint4*>(
          A + (size_t)(row0 + arow) * K + (k0 + BK) + acol);
      ra0 = gA[0]; ra1 = gA[1];
      const uint4* gB = reinterpret_cast<const uint4*>(
          Bm + (size_t)(k0 + BK + brow) * N + col0 + bcol);
      rb0 = gB[0]; rb1 = gB[1];
      if (k0 + 2 * BK < K) {   // prefetch tile k+2 into L2 (global_prefetch_b8)
        __builtin_prefetch(A + (size_t)(row0 + arow) * K + (k0 + 2 * BK) + acol, 0, 3);
        __builtin_prefetch(Bm + (size_t)(k0 + 2 * BK + brow) * N + col0 + bcol, 0, 3);
      }
    }

    // A fragments: lane half 0 -> K {0..7,16..23}; half 1 -> K {8..15,24..31}
    FragAB afrag[2];
#pragma unroll
    for (int mi = 0; mi < 2; ++mi) {
      const int r  = wm * 32 + mi * 16 + l16;
      const int kb = lh * 8;
      afrag[mi].q[0] = *reinterpret_cast<const uint4*>(&sA[buf][r][kb]);
      afrag[mi].q[1] = *reinterpret_cast<const uint4*>(&sA[buf][r][kb + 16]);
    }
    // B fragments: lane = N-column (l16); half 0 -> K {0..7,16..23}; half 1 -> K {8..15,24..31}
    FragAB bfrag[4];
#pragma unroll
    for (int ni = 0; ni < 4; ++ni) {
      const int cb = wn * 64 + ni * 16;
#pragma unroll
      for (int s = 0; s < 16; ++s) {
        const int kk = (s < 8) ? (lh * 8 + s) : (16 + lh * 8 + (s - 8));
        bfrag[ni].v[s] = sB[buf][kk][cb + l16];
      }
    }
#pragma unroll
    for (int mi = 0; mi < 2; ++mi)
#pragma unroll
      for (int ni = 0; ni < 4; ++ni)
        acc[mi][ni] = __builtin_amdgcn_wmma_f32_16x16x32_bf16(
            false, afrag[mi].v, false, bfrag[ni].v,
            (short)0, acc[mi][ni], false, false);

    if (has_next) {    // stage tile k+1 into the alternate buffer
      const int nb = buf ^ 1;
      *reinterpret_cast<uint4*>(&sA[nb][arow][acol])     = ra0;
      *reinterpret_cast<uint4*>(&sA[nb][arow][acol + 8]) = ra1;
      *reinterpret_cast<uint4*>(&sB[nb][brow][bcol])     = rb0;
      *reinterpret_cast<uint4*>(&sB[nb][brow][bcol + 8]) = rb1;
    }
    buf ^= 1;
  }

  // epilogue: C layout -> VGPR r holds rows {r, r+8} split by lane half
#pragma unroll
  for (int mi = 0; mi < 2; ++mi) {
#pragma unroll
    for (int ni = 0; ni < 4; ++ni) {
      const int col = col0 + wn * 64 + ni * 16 + l16;
      const float bv = bias[col];
#pragma unroll
      for (int r = 0; r < 8; ++r) {
        const int row = row0 + wm * 32 + mi * 16 + lh * 8 + r;
        const size_t oidx = (size_t)row * N + col;
        float v = acc[mi][ni][r] + bv;
        if (EPI == 0) {
          outB[oidx] = (bf16_t)v;
        } else if (EPI == 1) {
          v = 0.5f * v * (1.0f + erff(v * 0.70710678118654752f));
          outB[oidx] = (bf16_t)v;
        } else {
          outF[oidx] = v + resid[oidx];
        }
      }
    }
  }
}

// ---------------- window attention: one block per window, one wave per head ----------------
__global__ __launch_bounds__(256)
void attn_kernel(const bf16_t* __restrict__ qkv, const float* __restrict__ bias_table,
                 bf16_t* __restrict__ attn_out) {
  __shared__ float s_qkv[NTOK][3 * DIM];           // 55296 B
  __shared__ float s_sc[HEADS][NTOK][NTOK];        // 2592 B
  const int b   = blockIdx.x;
  const int tid = threadIdx.x;
  const bf16_t* pq = qkv + (size_t)b * NTOK * (3 * DIM);
  for (int i = tid; i < NTOK * 3 * DIM; i += 256)
    s_qkv[i / (3 * DIM)][i % (3 * DIM)] = (float)pq[i];
  __syncthreads();

  const int h    = tid >> 5;
  const int lane = tid & 31;
  const int qo = h * HEAD_DIM;
  const int ko = DIM + h * HEAD_DIM;
  const int vo = 2 * DIM + h * HEAD_DIM;

  // scores + relative position bias
  for (int e = lane; e < NTOK * NTOK; e += 32) {
    const int n = e / NTOK, m = e % NTOK;
    float s = 0.f;
#pragma unroll
    for (int d = 0; d < HEAD_DIM; ++d)
      s += s_qkv[n][qo + d] * s_qkv[m][ko + d];
    const int dr = n / WIN - m / WIN + (WIN - 1);
    const int dc = n % WIN - m % WIN + (WIN - 1);
    const int ridx = dr * (2 * WIN - 1) + dc;
    s_sc[h][n][m] = s * ATTN_SCALE + bias_table[ridx * HEADS + h];
  }
  __syncthreads();

  // softmax per row (lane n handles row n of its head)
  if (lane < NTOK) {
    const int n = lane;
    float mx = -1e30f;
#pragma unroll
    for (int m = 0; m < NTOK; ++m) mx = fmaxf(mx, s_sc[h][n][m]);
    float ex[NTOK], sum = 0.f;
#pragma unroll
    for (int m = 0; m < NTOK; ++m) { ex[m] = expf(s_sc[h][n][m] - mx); sum += ex[m]; }
    const float inv = 1.0f / sum;
#pragma unroll
    for (int m = 0; m < NTOK; ++m) s_sc[h][n][m] = ex[m] * inv;
  }
  __syncthreads();

  // out = attn @ v, written to (B, N, H*d) layout for the proj GEMM
  for (int e = lane; e < NTOK * HEAD_DIM; e += 32) {
    const int n = e / HEAD_DIM, d = e % HEAD_DIM;
    float s = 0.f;
#pragma unroll
    for (int m = 0; m < NTOK; ++m)
      s += s_sc[h][n][m] * s_qkv[m][vo + d];
    attn_out[((size_t)(b * NTOK + n)) * DIM + h * HEAD_DIM + d] = (bf16_t)s;
  }
}

// ---------------- depthwise 3x3 merge: y[b][c] = sum_n x2[b][n][c]*w[c][n] + b[c] ----------------
__global__ void convmerge_kernel(const float* __restrict__ x2, const float* __restrict__ wconv,
                                 const float* __restrict__ bconv, float* __restrict__ out) {
  const int idx = blockIdx.x * blockDim.x + threadIdx.x;
  if (idx >= NWIN * DIM) return;
  const int bw = idx / DIM, c = idx % DIM;
  float s = bconv[c];
#pragma unroll
  for (int n = 0; n < NTOK; ++n)
    s += x2[((size_t)(bw * NTOK + n)) * DIM + c] * wconv[c * NTOK + n];
  out[idx] = s;
}

// ---------------- orchestration ----------------
extern "C" void kernel_launch(void* const* d_in, const int* in_sizes, int n_in,
                              void* d_out, int out_size, void* d_ws, size_t ws_size,
                              hipStream_t stream) {
  (void)in_sizes; (void)n_in; (void)out_size; (void)ws_size;
  const float* x      = (const float*)d_in[0];
  const float* gamma1 = (const float*)d_in[1];
  const float* beta1  = (const float*)d_in[2];
  const float* w_qkv  = (const float*)d_in[3];
  const float* b_qkv  = (const float*)d_in[4];
  const float* btab   = (const float*)d_in[5];
  const float* w_proj = (const float*)d_in[6];
  const float* b_proj = (const float*)d_in[7];
  const float* gamma2 = (const float*)d_in[8];
  const float* beta2  = (const float*)d_in[9];
  const float* w_fc1  = (const float*)d_in[10];
  const float* b_fc1  = (const float*)d_in[11];
  const float* w_fc2  = (const float*)d_in[12];
  const float* b_fc2  = (const float*)d_in[13];
  const float* w_conv = (const float*)d_in[14];
  const float* b_conv = (const float*)d_in[15];
  float* out = (float*)d_out;

  char* ws = (char*)d_ws;
  size_t off = 0;
  auto carve = [&](size_t bytes) -> char* {
    char* p = ws + off;
    off = (off + bytes + 255) & ~(size_t)255;
    return p;
  };
  bf16_t* h1    = (bf16_t*)carve((size_t)TOKENS * DIM * 2);
  bf16_t* qkvB  = (bf16_t*)carve((size_t)TOKENS * 3 * DIM * 2);
  bf16_t* attnB = (bf16_t*)carve((size_t)TOKENS * DIM * 2);
  float*  x1    = (float*) carve((size_t)TOKENS * DIM * 4);
  bf16_t* h2    = (bf16_t*)carve((size_t)TOKENS * DIM * 2);
  bf16_t* hidB  = (bf16_t*)carve((size_t)TOKENS * HIDN * 2);
  float*  x2    = (float*) carve((size_t)TOKENS * DIM * 4);
  bf16_t* wqT   = (bf16_t*)carve((size_t)DIM * 3 * DIM * 2);
  bf16_t* wpT   = (bf16_t*)carve((size_t)DIM * DIM * 2);
  bf16_t* w1T   = (bf16_t*)carve((size_t)DIM * HIDN * 2);
  bf16_t* w2T   = (bf16_t*)carve((size_t)HIDN * DIM * 2);

  const dim3 blk(256);

  // weight conversions (W is [N_out, K_in] row-major -> bf16 [K][N])
  cvt_wT_kernel<<<(3 * DIM * DIM + 255) / 256, blk, 0, stream>>>(w_qkv, wqT, 3 * DIM, DIM);
  cvt_wT_kernel<<<(DIM * DIM + 255) / 256,     blk, 0, stream>>>(w_proj, wpT, DIM, DIM);
  cvt_wT_kernel<<<(HIDN * DIM + 255) / 256,    blk, 0, stream>>>(w_fc1, w1T, HIDN, DIM);
  cvt_wT_kernel<<<(DIM * HIDN + 255) / 256,    blk, 0, stream>>>(w_fc2, w2T, DIM, HIDN);

  // LN1
  ln_bf16_kernel<<<TOKENS / 8, blk, 0, stream>>>(x, gamma1, beta1, h1);

  // QKV GEMM: (73728 x 512) x (512 x 1536) +bias -> bf16
  gemm_wmma_kernel<0><<<dim3(3 * DIM / BN, TOKENS / BM), blk, 0, stream>>>(
      h1, wqT, b_qkv, nullptr, qkvB, nullptr, TOKENS, 3 * DIM, DIM);

  // window attention
  attn_kernel<<<NWIN, blk, 0, stream>>>(qkvB, btab, attnB);

  // proj GEMM +bias +residual(x) -> f32 x1
  gemm_wmma_kernel<2><<<dim3(DIM / BN, TOKENS / BM), blk, 0, stream>>>(
      attnB, wpT, b_proj, x, nullptr, x1, TOKENS, DIM, DIM);

  // LN2
  ln_bf16_kernel<<<TOKENS / 8, blk, 0, stream>>>(x1, gamma2, beta2, h2);

  // FC1 GEMM +bias +GELU -> bf16 hidden
  gemm_wmma_kernel<1><<<dim3(HIDN / BN, TOKENS / BM), blk, 0, stream>>>(
      h2, w1T, b_fc1, nullptr, hidB, nullptr, TOKENS, HIDN, DIM);

  // FC2 GEMM +bias +residual(x1) -> f32 x2
  gemm_wmma_kernel<2><<<dim3(DIM / BN, TOKENS / BM), blk, 0, stream>>>(
      hidB, w2T, b_fc2, x1, nullptr, x2, TOKENS, DIM, HIDN);

  // depthwise 3x3 merge -> (B, 1, C)
  convmerge_kernel<<<(NWIN * DIM + 255) / 256, blk, 0, stream>>>(x2, w_conv, b_conv, out);
}
